// Module_78460462564130
// MI455X (gfx1250) — matrix operand order repaired and checked by reading, not compiled, not run on hardware
//
#include <hip/hip_runtime.h>
#include <hip/hip_bf16.h>

// NAIS concat-attention scoring, restructured:
//   qc      = Q @ W_q + bias                  (per (b,h), tiny; folded into WMMA C-init)
//   hidden^T= W_k^T (64x64) @ K^T (64x16)     (WMMA f16, d-dim lands inside each lane)
//   score   = sum_d relu(hidden) * wo         (v_med3 relu + pk FMAs + ONE shfl_xor(16))
//
// B=32 H=8 KLEN=8192 D=64. HBM-bound (~536MB of K traffic ~ 23us at 23.3TB/s).

typedef __attribute__((ext_vector_type(16))) _Float16 v16h;
typedef __attribute__((ext_vector_type(8)))  float    v8f;

#define NB   32
#define NH   8
#define NK   8192
#define ND   64

#define CHUNKS   4            // blocks along k
#define ROWS_PB  (NK / CHUNKS)  // 2048 rows per block
#define ITERS    (ROWS_PB / 128) // 16 iterations (8 waves x 16 rows)

// relu in a single VALU op: v_med3_f32(x, 0, +inf)
__device__ __forceinline__ float relu1(float x) {
    return __builtin_amdgcn_fmed3f(x, 0.0f, __builtin_inff());
}

// Load 8 contiguous f32, convert to f16 into dst[base..base+7] (pairs -> v_cvt_pk_f16_f32).
__device__ __forceinline__ void cvt8(v16h& dst, int base, const float* p) {
    float4 x = ((const float4*)p)[0];
    float4 y = ((const float4*)p)[1];
    dst[base + 0] = (_Float16)x.x; dst[base + 1] = (_Float16)x.y;
    dst[base + 2] = (_Float16)x.z; dst[base + 3] = (_Float16)x.w;
    dst[base + 4] = (_Float16)y.x; dst[base + 5] = (_Float16)y.y;
    dst[base + 6] = (_Float16)y.z; dst[base + 7] = (_Float16)y.w;
}

// Load 16 contiguous f32 -> v16h (consecutive e values, pairwise packed as B-operand wants).
__device__ __forceinline__ v16h ldcvt16(const float* p) {
    v16h r;
    cvt8(r, 0, p);
    cvt8(r, 8, p + 8);
    return r;
}

__global__ __launch_bounds__(256)
void nais_score_kernel(const float* __restrict__ Q,
                       const float* __restrict__ K,
                       const float* __restrict__ Wc,
                       const float* __restrict__ Wo,
                       const float* __restrict__ bias,
                       float* __restrict__ out) {
    const int bh    = blockIdx.y;          // 0..B*H-1
    const int hh    = bh & (NH - 1);       // head index
    const int chunk = blockIdx.x;          // 0..CHUNKS-1
    const int tid   = threadIdx.x;
    const int lane  = tid & 31;
    const int wave  = tid >> 5;            // 0..7
    const int lhalf = lane >> 4;           // 0/1
    const int lmod  = lane & 15;

    __shared__ float s_qc[ND];
    __shared__ float s_wo[ND];

    // ---- qc = Q @ W_q + bias ; wo broadcast (threads 0..63) ----
    if (tid < ND) {
        const int d = tid;
        float acc = bias[hh * ND + d];
        const float* q  = Q  + (size_t)bh * ND;               // [b,h,1,64]
        const float* wq = Wc + (size_t)hh * 2 * ND * ND + d;  // W_c[h][e][d], e-stride 64
        #pragma unroll 8
        for (int e = 0; e < ND; ++e) acc += q[e] * wq[e * ND];
        s_qc[d] = acc;
        s_wo[d] = Wo[hh * ND + d];                            // [h,64,1]
    }
    __syncthreads();

    // ---- A = W_k^T tiles in f16 registers ----
    // A 16x32 f16 layout: lane l: M = l%16 (=d_local), halves 0..7 hold
    // e = es*32 + lhalf*16 + i, halves 8..15 hold e = es*32 + lhalf*16 + 8 + i.
    const float* wk = Wc + ((size_t)hh * 2 * ND + ND) * ND;   // W_k[e][d], e-stride 64
    v16h Av[4][2];
    #pragma unroll
    for (int t = 0; t < 4; ++t) {
        const int m = t * 16 + lmod;                          // d row of W_k^T
        #pragma unroll
        for (int es = 0; es < 2; ++es) {
            #pragma unroll
            for (int i = 0; i < 8; ++i) {
                const int e_lo = es * 32 + lhalf * 16 + i;
                const int e_hi = e_lo + 8;
                Av[t][es][i]     = (_Float16)wk[e_lo * ND + m];
                Av[t][es][8 + i] = (_Float16)wk[e_hi * ND + m];
            }
        }
    }

    // ---- C-init with qc[d] (broadcast along N) and wo[d] registers ----
    // D element (M=d, N=krow): lane l, VGPR j -> d = t*16 + lhalf*8 + j.
    v8f  cinit[4];
    float wov[4][8];
    #pragma unroll
    for (int t = 0; t < 4; ++t) {
        #pragma unroll
        for (int j = 0; j < 8; ++j) {
            const int d = t * 16 + lhalf * 8 + j;
            cinit[t][j] = s_qc[d];
            wov[t][j]   = s_wo[d];
        }
    }

    const float* Kbh = K   + (size_t)bh * NK * ND;
    float*       obh = out + (size_t)bh * NK;

    const int rowchunk = chunk * ROWS_PB;
    for (int it = 0; it < ITERS; ++it) {
        const int rowbase = rowchunk + it * 128 + wave * 16;
        // B = K^T tile: lane's k-row = rowbase + lmod; e chunk = es*32 + lhalf*16 + [0..16)
        const float* bp = Kbh + (size_t)(rowbase + lmod) * ND + lhalf * 16;

        if (it < ITERS - 1) __builtin_prefetch(bp + 128 * ND, 0, 1);  // next iter's rows

        v16h b0 = ldcvt16(bp);        // e-step 0
        v16h b1 = ldcvt16(bp + 32);   // e-step 1

        v8f c0, c1, c2, c3;
        c0 = __builtin_amdgcn_wmma_f32_16x16x32_f16(false, Av[0][0], false, b0, (short)0, cinit[0], false, false);
        c1 = __builtin_amdgcn_wmma_f32_16x16x32_f16(false, Av[1][0], false, b0, (short)0, cinit[1], false, false);
        c2 = __builtin_amdgcn_wmma_f32_16x16x32_f16(false, Av[2][0], false, b0, (short)0, cinit[2], false, false);
        c3 = __builtin_amdgcn_wmma_f32_16x16x32_f16(false, Av[3][0], false, b0, (short)0, cinit[3], false, false);
        c0 = __builtin_amdgcn_wmma_f32_16x16x32_f16(false, Av[0][1], false, b1, (short)0, c0, false, false);
        c1 = __builtin_amdgcn_wmma_f32_16x16x32_f16(false, Av[1][1], false, b1, (short)0, c1, false, false);
        c2 = __builtin_amdgcn_wmma_f32_16x16x32_f16(false, Av[2][1], false, b1, (short)0, c2, false, false);
        c3 = __builtin_amdgcn_wmma_f32_16x16x32_f16(false, Av[3][1], false, b1, (short)0, c3, false, false);

        // ---- epilogue: single-op relu + dot(wo), all in-lane; one cross-half combine ----
        float s0 = 0.f, s1 = 0.f, s2 = 0.f, s3 = 0.f;
        #pragma unroll
        for (int j = 0; j < 8; ++j) {
            s0 = fmaf(relu1(c0[j]), wov[0][j], s0);
            s1 = fmaf(relu1(c1[j]), wov[1][j], s1);
            s2 = fmaf(relu1(c2[j]), wov[2][j], s2);
            s3 = fmaf(relu1(c3[j]), wov[3][j], s3);
        }
        float s = (s0 + s1) + (s2 + s3);
        s += __shfl_xor(s, 16, 32);      // combine the two d-halves
        if (lane < 16) obh[rowbase + lane] = s;   // 16 consecutive f32 per wave
    }
}

extern "C" void kernel_launch(void* const* d_in, const int* in_sizes, int n_in,
                              void* d_out, int out_size, void* d_ws, size_t ws_size,
                              hipStream_t stream) {
    (void)in_sizes; (void)n_in; (void)out_size; (void)d_ws; (void)ws_size;
    const float* Q    = (const float*)d_in[0];
    const float* K    = (const float*)d_in[1];
    const float* Wc   = (const float*)d_in[2];
    const float* Wo   = (const float*)d_in[3];
    const float* bias = (const float*)d_in[4];
    float* out = (float*)d_out;

    dim3 grid(CHUNKS, NB * NH);      // 4 x 256 = 1024 blocks
    dim3 block(256);                  // 8 waves (wave32)
    nais_score_kernel<<<grid, block, 0, stream>>>(Q, K, Wc, Wo, bias, out);
}
